// attention_10574209482773
// MI455X (gfx1250) — matrix operand order repaired and checked by reading, not compiled, not run on hardware
//
#include <hip/hip_runtime.h>

// ---------------- Types / WMMA helpers (gfx1250 wave32) ----------------
typedef __attribute__((ext_vector_type(16))) __bf16 v16bf;
typedef __attribute__((ext_vector_type(8)))  __bf16 v8bf;
typedef __attribute__((ext_vector_type(8)))  float  v8f;

union V16U { v16bf v; v8bf h[2]; };

__device__ __forceinline__ v16bf load_frag_2x8(const __bf16* p0, const __bf16* p1) {
    V16U u;
    u.h[0] = *(const v8bf*)p0;
    u.h[1] = *(const v8bf*)p1;
    return u.v;
}

// A-fragment (16x32 bf16) from row-major src with leading dim ld.
// ISA layout: lane r=L&15 is row; K halves {kb..kb+7} and {16+kb..16+kb+7}, kb=(L>=16)*8.
__device__ __forceinline__ v16bf load_a_frag(const __bf16* src, int ld) {
    int lane = threadIdx.x & 31;
    int r    = lane & 15;
    int kb   = (lane >> 4) << 3;             // 0 or 8
    const __bf16* p = src + r * ld + kb;
    return load_frag_2x8(p, p + 16);
}

// B-fragment (32x16 bf16): K halves {kb..kb+7} and {16+kb..16+kb+7}.  srcT is "n-major": srcT[n][k], ld = row length.
// lane n=L&15 picks column; kb=(L>=16)*8; two b128 loads (same K order as A).
__device__ __forceinline__ v16bf load_b_frag(const __bf16* srcT, int ld) {
    int lane = threadIdx.x & 31;
    int n    = lane & 15;
    int kb   = (lane >> 4) << 3;             // 0 or 8
    const __bf16* p = srcT + n * ld + kb;
    return load_frag_2x8(p, p + 16);
}

__device__ __forceinline__ v8f wmma_bf16(v16bf a, v16bf b, v8f c) {
    return __builtin_amdgcn_wmma_f32_16x16x32_bf16(
        /*neg_a=*/false, a, /*neg_b=*/false, b,
        /*c_mod=*/(short)0, c, /*reuse_a=*/false, /*reuse_b=*/false);
}

// ---------------- Problem constants ----------------
#define BATCH   2
#define SEQ     2048
#define DMODEL  1024
#define NHEAD   16
#define HDIM    64
#define MROWS   (BATCH * SEQ)      // 4096

// ---------------- Conversion kernels ----------------
__global__ void cvt_f32_to_bf16(const float* __restrict__ in, __bf16* __restrict__ out, int n) {
    int i = blockIdx.x * blockDim.x + threadIdx.x;
    if (i < n) out[i] = (__bf16)in[i];
}

// Wt[n][k] = W[k][n]  (W is [K=1024][N=1024] row-major)
__global__ void transpose_f32_to_bf16(const float* __restrict__ W, __bf16* __restrict__ Wt) {
    int i = blockIdx.x * blockDim.x + threadIdx.x;      // over 1024*1024
    int n = i >> 10;
    int k = i & 1023;
    Wt[i] = (__bf16)W[k * DMODEL + n];
}

// ---------------- Projection GEMM ----------------
// C[M=4096,N=1024] = A[M,K=1024] @ B[K,N], B given transposed (Bt[n][k]).
// Block: 128 threads / 4 waves; block tile 128(M) x 64(N); wave tile 32 x 64.
// Weight slice (64n x 32k bf16 = 4 KB) is double-buffered in LDS via
// global_load_async_to_lds_b128 and pipelined against the WMMAs.
// MODE 0: store bf16 head-split [B,H,S,64]
// MODE 1: store bf16 head-split transposed [B,H,64,S]   (for V)
// MODE 2: store f32 + bias to out (final projection)
template <int MODE>
__global__ void __launch_bounds__(128)
proj_gemm(const __bf16* __restrict__ A, const __bf16* __restrict__ Bt,
          void* __restrict__ outp, const float* __restrict__ bias) {
    const int K = DMODEL;
    __shared__ __align__(16) __bf16 Bs[2][64][32];   // double-buffered weight tile

    int w    = threadIdx.x >> 5;
    int lane = threadIdx.x & 31;
    int m0   = blockIdx.y * 128 + w * 32;
    int n0   = blockIdx.x * 64;

    // This thread's two 16-byte chunks of the 4 KB weight tile (256 chunks / 128 threads).
    int c0  = (int)threadIdx.x * 2;
    int bn0 = c0 >> 2,       bk0 = (c0 & 3) * 8;
    int bn1 = (c0 + 1) >> 2, bk1 = ((c0 + 1) & 3) * 8;

    auto issue_tile = [&](int buf, int k) {
        unsigned l0 = (unsigned)(size_t)&Bs[buf][bn0][bk0];
        unsigned l1 = (unsigned)(size_t)&Bs[buf][bn1][bk1];
        const __bf16* g0 = Bt + (size_t)(n0 + bn0) * K + k + bk0;
        const __bf16* g1 = Bt + (size_t)(n0 + bn1) * K + k + bk1;
        asm volatile("global_load_async_to_lds_b128 %0, %1, off"
                     :: "v"(l0), "v"(g0) : "memory");
        asm volatile("global_load_async_to_lds_b128 %0, %1, off"
                     :: "v"(l1), "v"(g1) : "memory");
    };

    v8f acc[2][4];
#pragma unroll
    for (int mi = 0; mi < 2; ++mi)
#pragma unroll
        for (int j = 0; j < 4; ++j)
#pragma unroll
            for (int i = 0; i < 8; ++i) acc[mi][j][i] = 0.0f;

    issue_tile(0, 0);
    for (int k = 0, step = 0; k < K; k += 32, ++step) {
        int cur = step & 1;
        if (k + 32 < K) {
            issue_tile(cur ^ 1, k + 32);
            // 2 ops for the next tile stay in flight; async loads complete in order,
            // so <=2 releases exactly the current tile.
            asm volatile("s_wait_asynccnt 0x2" ::: "memory");
        } else {
            asm volatile("s_wait_asynccnt 0x0" ::: "memory");
        }
        __syncthreads();                          // current tile visible to all waves

        v16bf a0 = load_a_frag(A + (size_t)m0 * K + k, K);
        v16bf a1 = load_a_frag(A + (size_t)(m0 + 16) * K + k, K);
#pragma unroll
        for (int j = 0; j < 4; ++j) {
            v16bf b = load_b_frag(&Bs[cur][j * 16][0], 32);
            acc[0][j] = wmma_bf16(a0, b, acc[0][j]);
            acc[1][j] = wmma_bf16(a1, b, acc[1][j]);
        }
        __syncthreads();                          // all reads done before buffer reuse
    }

    int hl = lane >> 4;           // C-layout: row = i + 8*hl, col = lane&15
    int nl = lane & 15;
#pragma unroll
    for (int mi = 0; mi < 2; ++mi) {
#pragma unroll
        for (int j = 0; j < 4; ++j) {
#pragma unroll
            for (int i = 0; i < 8; ++i) {
                int n = n0 + j * 16 + nl;
                int m = m0 + mi * 16 + i + 8 * hl;
                if (MODE == 2) {
                    ((float*)outp)[(size_t)m * DMODEL + n] = acc[mi][j][i] + bias[n];
                } else {
                    int b = m >> 11, s = m & (SEQ - 1);
                    int h = n >> 6,  d = n & (HDIM - 1);
                    __bf16* dst = (__bf16*)outp;
                    if (MODE == 0)
                        dst[(((size_t)(b * NHEAD + h) * SEQ) + s) * HDIM + d] = (__bf16)acc[mi][j][i];
                    else
                        dst[(((size_t)(b * NHEAD + h) * HDIM) + d) * SEQ + s] = (__bf16)acc[mi][j][i];
                }
            }
        }
    }
}

// ---------------- Flash attention (causal, mask added before 1/sqrt(Dh) scale) ----------------
// Qh,Kh: [B,H,S,64] bf16 ; Vt: [B,H,64,S] bf16 ; ctx: [B,S,1024] bf16
__global__ void __launch_bounds__(128)
attn_fwd(const __bf16* __restrict__ Qh, const __bf16* __restrict__ Kh,
         const __bf16* __restrict__ Vt, __bf16* __restrict__ ctx) {
    __shared__ __align__(16) __bf16 P[4][16][32];   // per-wave probability tile (C->A layout bounce)

    int w    = threadIdx.x >> 5;
    int lane = threadIdx.x & 31;
    int nl   = lane & 15;
    int hl   = lane >> 4;

    const int qtiles = SEQ / 64;
    int t  = blockIdx.x % qtiles;
    int bh = blockIdx.x / qtiles;
    int b  = bh / NHEAD;
    int h  = bh % NHEAD;
    int q0 = t * 64 + w * 16;

    const __bf16* Qbase = Qh + (size_t)bh * SEQ * HDIM + (size_t)q0 * HDIM;
    const __bf16* Kbase = Kh + (size_t)bh * SEQ * HDIM;
    const __bf16* Vbase = Vt + (size_t)bh * HDIM * SEQ;

    // Q fragments (dims 0..31 and 32..63) held for the whole kernel
    v16bf qf0 = load_a_frag(Qbase, HDIM);
    v16bf qf1 = load_a_frag(Qbase + 32, HDIM);

    v8f o[4];
    float m_i[8], l_i[8];
#pragma unroll
    for (int j = 0; j < 4; ++j)
#pragma unroll
        for (int i = 0; i < 8; ++i) o[j][i] = 0.0f;
#pragma unroll
    for (int i = 0; i < 8; ++i) { m_i[i] = -1e30f; l_i[i] = 0.0f; }

    const float SCALE = 0.125f;              // 1/sqrt(64)
    int ktmax = (q0 + 15) >> 5;              // last 32-key block with any unmasked entry

    for (int kt = 0; kt <= ktmax; ++kt) {
        int key0 = kt * 32;
        if (kt < ktmax)
            __builtin_prefetch(Kbase + (size_t)(key0 + 32) * HDIM, 0, 3);

        // ---- scores: S = Q @ K^T (two 16-key tiles) ----
        v8f s0, s1;
#pragma unroll
        for (int i = 0; i < 8; ++i) { s0[i] = 0.0f; s1[i] = 0.0f; }
        s0 = wmma_bf16(qf0, load_b_frag(Kbase + (size_t)key0 * HDIM,             HDIM), s0);
        s0 = wmma_bf16(qf1, load_b_frag(Kbase + (size_t)key0 * HDIM + 32,        HDIM), s0);
        s1 = wmma_bf16(qf0, load_b_frag(Kbase + (size_t)(key0 + 16) * HDIM,      HDIM), s1);
        s1 = wmma_bf16(qf1, load_b_frag(Kbase + (size_t)(key0 + 16) * HDIM + 32, HDIM), s1);

        // ---- masked online softmax update ----
        float corr[8];
#pragma unroll
        for (int i = 0; i < 8; ++i) {
            int qrow = q0 + i + 8 * hl;
            float z0 = (s0[i] + ((key0 + nl      > qrow) ? -1e9f : 0.0f)) * SCALE;
            float z1 = (s1[i] + ((key0 + 16 + nl > qrow) ? -1e9f : 0.0f)) * SCALE;
            float mx = fmaxf(z0, z1);
            mx = fmaxf(mx, __shfl_xor(mx, 1, 32));
            mx = fmaxf(mx, __shfl_xor(mx, 2, 32));
            mx = fmaxf(mx, __shfl_xor(mx, 4, 32));
            mx = fmaxf(mx, __shfl_xor(mx, 8, 32));
            float mnew = fmaxf(m_i[i], mx);
            float c  = __expf(m_i[i] - mnew);
            float p0 = __expf(z0 - mnew);
            float p1 = __expf(z1 - mnew);
            float rs = p0 + p1;
            rs += __shfl_xor(rs, 1, 32);
            rs += __shfl_xor(rs, 2, 32);
            rs += __shfl_xor(rs, 4, 32);
            rs += __shfl_xor(rs, 8, 32);
            l_i[i] = l_i[i] * c + rs;
            m_i[i] = mnew;
            corr[i] = c;
            // C-layout -> LDS (row = i + 8*hl, cols nl and nl+16), wave-private, in-order DS
            P[w][i + 8 * hl][nl]      = (__bf16)p0;
            P[w][i + 8 * hl][nl + 16] = (__bf16)p1;
        }

        // rescale running output
#pragma unroll
        for (int j = 0; j < 4; ++j)
#pragma unroll
            for (int i = 0; i < 8; ++i) o[j][i] *= corr[i];

        // ---- P @ V ----
        v16bf pf = load_a_frag(&P[w][0][0], 32);
#pragma unroll
        for (int j = 0; j < 4; ++j) {
            v16bf bv = load_b_frag(Vbase + (size_t)(j * 16) * SEQ + key0, SEQ);
            o[j] = wmma_bf16(pf, bv, o[j]);
        }
    }

    // ---- finalize: divide by l, store merged-head bf16 context ----
#pragma unroll
    for (int i = 0; i < 8; ++i) l_i[i] = 1.0f / l_i[i];
#pragma unroll
    for (int j = 0; j < 4; ++j) {
#pragma unroll
        for (int i = 0; i < 8; ++i) {
            int srow = q0 + i + 8 * hl;
            ctx[((size_t)(b * SEQ + srow)) * DMODEL + h * HDIM + j * 16 + nl] =
                (__bf16)(o[j][i] * l_i[i]);
        }
    }
}

// ---------------- Host launch ----------------
extern "C" void kernel_launch(void* const* d_in, const int* in_sizes, int n_in,
                              void* d_out, int out_size, void* d_ws, size_t ws_size,
                              hipStream_t stream) {
    const float* keys    = (const float*)d_in[0];
    const float* queries = (const float*)d_in[1];
    const float* values  = (const float*)d_in[2];
    const float* Wq      = (const float*)d_in[3];
    const float* Wk      = (const float*)d_in[4];
    const float* Wv      = (const float*)d_in[5];
    const float* Wf      = (const float*)d_in[6];
    const float* bf      = (const float*)d_in[7];

    const size_t ACT = (size_t)MROWS * DMODEL;   // 4096*1024 elements
    const size_t WEL = (size_t)DMODEL * DMODEL;  // 1024*1024 elements

    char* ws = (char*)d_ws;
    __bf16* qb  = (__bf16*)ws;                  // activations bf16 (8 MB each)
    __bf16* kb  = qb + ACT;
    __bf16* vb  = kb + ACT;
    __bf16* WqT = vb + ACT;                     // transposed bf16 weights (2 MB each)
    __bf16* WkT = WqT + WEL;
    __bf16* WvT = WkT + WEL;
    __bf16* WfT = WvT + WEL;
    __bf16* Qh  = WfT + WEL;                    // [B,H,S,64] (8 MB each)
    __bf16* Kh  = Qh + ACT;
    __bf16* Vt  = Kh + ACT;                     // [B,H,64,S]
    __bf16* ctxb = Vt + ACT;                    // [B,S,1024]

    // 1) convert activations to bf16
    int cvtBlocks = (int)((ACT + 255) / 256);
    cvt_f32_to_bf16<<<cvtBlocks, 256, 0, stream>>>(queries, qb, (int)ACT);
    cvt_f32_to_bf16<<<cvtBlocks, 256, 0, stream>>>(keys,    kb, (int)ACT);
    cvt_f32_to_bf16<<<cvtBlocks, 256, 0, stream>>>(values,  vb, (int)ACT);

    // 2) transpose+convert weights
    int trBlocks = (int)((WEL + 255) / 256);
    transpose_f32_to_bf16<<<trBlocks, 256, 0, stream>>>(Wq, WqT);
    transpose_f32_to_bf16<<<trBlocks, 256, 0, stream>>>(Wk, WkT);
    transpose_f32_to_bf16<<<trBlocks, 256, 0, stream>>>(Wv, WvT);
    transpose_f32_to_bf16<<<trBlocks, 256, 0, stream>>>(Wf, WfT);

    // 3) Q/K/V projections (WMMA bf16, async-LDS-staged weights)
    dim3 ggrid(DMODEL / 64, MROWS / 128);       // (16, 32)
    proj_gemm<0><<<ggrid, 128, 0, stream>>>(qb, WqT, (void*)Qh, nullptr);
    proj_gemm<0><<<ggrid, 128, 0, stream>>>(kb, WkT, (void*)Kh, nullptr);
    proj_gemm<1><<<ggrid, 128, 0, stream>>>(vb, WvT, (void*)Vt, nullptr);

    // 4) causal flash attention
    attn_fwd<<<BATCH * NHEAD * (SEQ / 64), 128, 0, stream>>>(Qh, Kh, Vt, ctxb);

    // 5) output projection + bias (f32 out)
    proj_gemm<2><<<ggrid, 128, 0, stream>>>(ctxb, WfT, d_out, bf);
}
